// DEBertALayer_81080392614803
// MI455X (gfx1250) — compile-verified, matrix operand order checked by reading, not run on hardware
//
#include <hip/hip_runtime.h>
#include <math.h>

// ---------------- problem constants ----------------
#define BATCH 2
#define SEQ   2048
#define EMB   1024
#define HEADS 16
#define HDIM  64
#define MROWS (BATCH * SEQ)          // 4096
#define FF    (4 * EMB)              // 4096

typedef __attribute__((ext_vector_type(16))) __bf16 v16bf;
typedef __attribute__((ext_vector_type(8)))  __bf16 bf16x8;
typedef __attribute__((ext_vector_type(4)))  __bf16 bf16x4;
typedef __attribute__((ext_vector_type(8)))  float  v8f;

__device__ __forceinline__ __bf16 f2bf(float f) {
  union { float f; unsigned u; } c; c.f = f;
  unsigned r = c.u + 0x7FFFu + ((c.u >> 16) & 1u);   // round-to-nearest-even
  unsigned short h = (unsigned short)(r >> 16);
  return __builtin_bit_cast(__bf16, h);
}

// Build a v16bf WMMA fragment from two 16-byte LDS/global chunks.
// Matches CDNA5 16-bit A/B layout: elements 0..7 = K(half*8 + 0..7),
// elements 8..15 = K(16 + half*8 + 0..7).
__device__ __forceinline__ v16bf make_frag(const __bf16* p0, const __bf16* p1) {
  bf16x8 a = *(const bf16x8*)p0;
  bf16x8 b = *(const bf16x8*)p1;
  v16bf r;
#pragma unroll
  for (int i = 0; i < 8; ++i) { r[i] = a[i]; r[i + 8] = b[i]; }
  return r;
}

__device__ __forceinline__ v8f zero8() {
  v8f z;
#pragma unroll
  for (int i = 0; i < 8; ++i) z[i] = 0.0f;
  return z;
}

// ---- CDNA5 async global->LDS copy (ASYNCcnt-tracked DMA, bypasses VGPRs) ----
// Low 32 bits of a generic pointer to LDS are the LDS byte offset
// (ISA 10.2: LDS aperture).
__device__ __forceinline__ void async_load_b128(void* lds_dst, const void* gsrc) {
  unsigned lds_off = (unsigned)(unsigned long long)lds_dst;
  asm volatile("global_load_async_to_lds_b128 %0, %1, off"
               :: "v"(lds_off), "v"(gsrc) : "memory");
}
__device__ __forceinline__ void wait_async0() {
  asm volatile("s_wait_asynccnt 0x0" ::: "memory");
}

// ---------------- elementwise converts ----------------
__global__ void cvt_f32_bf16(const float* __restrict__ in, __bf16* __restrict__ out, long n) {
  long i = (long)blockIdx.x * blockDim.x + threadIdx.x;
  long stride = (long)gridDim.x * blockDim.x;
  for (; i < n; i += stride) out[i] = f2bf(in[i]);
}

// [H,E,D] -> [E, H*D] row-major bf16
__global__ void cvt_qkv_w(const float* __restrict__ in, __bf16* __restrict__ out) {
  long n = (long)EMB * EMB;
  long i = (long)blockIdx.x * blockDim.x + threadIdx.x;
  long stride = (long)gridDim.x * blockDim.x;
  for (; i < n; i += stride) {
    int e = (int)(i / EMB);
    int c = (int)(i % EMB);
    int h = c >> 6, d = c & 63;
    out[i] = f2bf(in[((long)h * EMB + e) * HDIM + d]);
  }
}

// ---------------- LayerNorm (row = B*S, E = 1024) ----------------
__global__ __launch_bounds__(256)
void ln_kernel(const float* __restrict__ x, const float* __restrict__ g,
               const float* __restrict__ be, float* __restrict__ outF,
               __bf16* __restrict__ outB) {
  const int row = blockIdx.x;
  const int t = threadIdx.x;
  const float4 v = ((const float4*)(x + (size_t)row * EMB))[t];
  float s1 = v.x + v.y + v.z + v.w;
  float s2 = v.x * v.x + v.y * v.y + v.z * v.z + v.w * v.w;
#pragma unroll
  for (int off = 16; off > 0; off >>= 1) {
    s1 += __shfl_xor(s1, off, 32);
    s2 += __shfl_xor(s2, off, 32);
  }
  __shared__ float sm1[8], sm2[8];
  const int wid = t >> 5, lane = t & 31;
  if (lane == 0) { sm1[wid] = s1; sm2[wid] = s2; }
  __syncthreads();
  float t1 = 0.f, t2 = 0.f;
#pragma unroll
  for (int i = 0; i < 8; ++i) { t1 += sm1[i]; t2 += sm2[i]; }
  const float mean = t1 * (1.0f / EMB);
  const float var  = t2 * (1.0f / EMB) - mean * mean;
  const float inv  = rsqrtf(var + 1e-5f);
  const float4 gg = ((const float4*)g)[t];
  const float4 bb = ((const float4*)be)[t];
  float4 o;
  o.x = (v.x - mean) * inv * gg.x + bb.x;
  o.y = (v.y - mean) * inv * gg.y + bb.y;
  o.z = (v.z - mean) * inv * gg.z + bb.z;
  o.w = (v.w - mean) * inv * gg.w + bb.w;
  ((float4*)(outF + (size_t)row * EMB))[t] = o;
  bf16x4 ob; ob[0] = f2bf(o.x); ob[1] = f2bf(o.y); ob[2] = f2bf(o.z); ob[3] = f2bf(o.w);
  *(bf16x4*)(outB + (size_t)row * EMB + t * 4) = ob;
}

// ---------------- generic bf16 GEMM with fused epilogue ----------------
// C[M,N] = A[M,K] * B[K,N] + bias ; optional exact GELU; optional fp32 residual.
// Block tile 128x128x64; 8 waves, each wave 64(M) x 32(N) = 4x2 WMMA tiles,
// 16 WMMAs per K-step. A tile staged via async global->LDS DMA.
#define BM 128
#define BN 128
#define BK 64
#define LDSTR (BK + 8)   // 72 bf16 = 144B rows, 16B aligned

__global__ __launch_bounds__(256)
void gemm_bf16(const __bf16* __restrict__ A, const __bf16* __restrict__ Bm,
               const float* __restrict__ bias, const float* __restrict__ residual,
               float* __restrict__ outF, __bf16* __restrict__ outB,
               int M, int N, int K, int act) {
  __shared__ __align__(16) __bf16 lsA[BM][LDSTR];
  __shared__ __align__(16) __bf16 lsB[BN][LDSTR];   // transposed: [n][k]

  const int t = threadIdx.x;
  const int wid = t >> 5, lane = t & 31;
  const int wm = wid & 1;        // 0..1 -> 64 rows each
  const int wn = wid >> 1;       // 0..3 -> 32 cols each
  const int lr = lane & 15, half = lane >> 4;
  const int bm0 = blockIdx.x * BM;
  const int bn0 = blockIdx.y * BN;

  v8f acc[4][2];
#pragma unroll
  for (int i = 0; i < 4; ++i)
#pragma unroll
    for (int j = 0; j < 2; ++j) acc[i][j] = zero8();

  const int ar = t >> 1, aseg = (t & 1) * 32;         // A: row, 32-elem segment
  const int bk = t >> 2, bn32 = (t & 3) * 32;         // B: k-row, 32-col segment

  for (int k0 = 0; k0 < K; k0 += BK) {
    __syncthreads();
    { // stage A tile (row-major) via async DMA: 4 x b128 per thread
      const __bf16* src = A + (size_t)(bm0 + ar) * K + k0 + aseg;
      __bf16* dst = &lsA[ar][aseg];
#pragma unroll
      for (int i = 0; i < 4; ++i)
        async_load_b128(dst + i * 8, src + i * 8);
    }
    { // stage B tile transposed into [n][k]
      const __bf16* src = Bm + (size_t)(k0 + bk) * N + bn0 + bn32;
      bf16x8 x[4];
#pragma unroll
      for (int s = 0; s < 4; ++s) x[s] = *(const bf16x8*)(src + s * 8);
#pragma unroll
      for (int s = 0; s < 4; ++s)
#pragma unroll
        for (int i = 0; i < 8; ++i)
          lsB[bn32 + s * 8 + i][bk] = x[s][i];
    }
    if (k0 + BK < K) {  // global_prefetch_b8 of next tiles
      __builtin_prefetch(A + (size_t)(bm0 + ar) * K + k0 + BK + aseg, 0, 1);
      __builtin_prefetch(Bm + (size_t)(k0 + BK + bk) * N + bn0 + bn32, 0, 1);
    }
    wait_async0();      // A-tile DMA complete (ASYNCcnt not fenced by barrier)
    __syncthreads();    // B-tile ds stores fenced here (dscnt)

#pragma unroll
    for (int f = 0; f < 2; ++f) {   // two K=32 sub-steps
      const int kf = f * 32 + half * 8;
      v16bf bfr[2];
#pragma unroll
      for (int nt = 0; nt < 2; ++nt) {
        const __bf16* p = &lsB[wn * 32 + nt * 16 + lr][kf];
        bfr[nt] = make_frag(p, p + 16);
      }
#pragma unroll
      for (int mt = 0; mt < 4; ++mt) {
        const __bf16* p = &lsA[wm * 64 + mt * 16 + lr][kf];
        v16bf afr = make_frag(p, p + 16);
#pragma unroll
        for (int nt = 0; nt < 2; ++nt)
          acc[mt][nt] = __builtin_amdgcn_wmma_f32_16x16x32_bf16(
              false, afr, false, bfr[nt], (short)0, acc[mt][nt], false, false);
      }
    }
  }

  // epilogue: bias -> activation -> residual -> store
#pragma unroll
  for (int mt = 0; mt < 4; ++mt)
#pragma unroll
    for (int nt = 0; nt < 2; ++nt) {
      const int gn  = bn0 + wn * 32 + nt * 16 + lr;
      const int gm0 = bm0 + wm * 64 + mt * 16 + half * 8;
      const float bv = bias ? bias[gn] : 0.0f;
#pragma unroll
      for (int j = 0; j < 8; ++j) {
        const size_t idx = (size_t)(gm0 + j) * N + gn;
        float v = acc[mt][nt][j] + bv;
        if (act == 1) v = 0.5f * v * (1.0f + erff(v * 0.70710678118654752f));
        if (residual) v += residual[idx];
        if (outF) outF[idx] = v;
        if (outB) outB[idx] = f2bf(v);
      }
    }
}

// ---------------- flash attention ----------------
// grid: (S/64, H, B), 128 threads = 4 waves; each wave owns 16 query rows.
__global__ __launch_bounds__(128)
void attn_kernel(const __bf16* __restrict__ Q, const __bf16* __restrict__ Kb,
                 const __bf16* __restrict__ Vb, const int* __restrict__ mask,
                 __bf16* __restrict__ ctx) {
  __shared__ __align__(16) __bf16 vT[HDIM][40];       // [d][key], 32-key block
  __shared__ __align__(16) __bf16 pS[4][16][40];      // per-wave P tile [m][key]

  const int t = threadIdx.x, w = t >> 5, lane = t & 31;
  const int lr = lane & 15, half = lane >> 4;
  const int b = blockIdx.z, h = blockIdx.y;
  const int q0 = blockIdx.x * 64 + w * 16;

  // Q A-fragments (16 x 64, two K=32 fragments), loaded once from global
  const size_t qrow = (size_t)(b * SEQ + q0 + lr) * EMB + h * HDIM;
  v16bf qA[2];
#pragma unroll
  for (int f = 0; f < 2; ++f) {
    const __bf16* p = Q + qrow + f * 32 + half * 8;
    qA[f] = make_frag(p, p + 16);
  }

  float mrun[8], lrun[8];
  v8f accv[4];
#pragma unroll
  for (int j = 0; j < 8; ++j) { mrun[j] = -1e30f; lrun[j] = 0.0f; }
#pragma unroll
  for (int i = 0; i < 4; ++i) accv[i] = zero8();

  const int vkey = t >> 2, vdg = (t & 3) * 16;  // V staging coords

  for (int kb = 0; kb < SEQ; kb += 32) {
    __syncthreads();
    { // stage V^T block: vT[d][key]
      const __bf16* src = Vb + (size_t)(b * SEQ + kb + vkey) * EMB + h * HDIM + vdg;
      bf16x8 x0 = *(const bf16x8*)src;
      bf16x8 x1 = *(const bf16x8*)(src + 8);
#pragma unroll
      for (int i = 0; i < 8; ++i) {
        vT[vdg + i][vkey]     = x0[i];
        vT[vdg + 8 + i][vkey] = x1[i];
      }
    }

    // scores: two 16x16 tiles (32 keys), K-fragments straight from global
    v8f sc[2];
#pragma unroll
    for (int nt = 0; nt < 2; ++nt) {
      const int key = kb + nt * 16 + lr;
      const __bf16* kp = Kb + (size_t)(b * SEQ + key) * EMB + h * HDIM;
      v8f a = zero8();
#pragma unroll
      for (int f = 0; f < 2; ++f) {
        v16bf kB = make_frag(kp + f * 32 + half * 8, kp + f * 32 + half * 8 + 16);
        a = __builtin_amdgcn_wmma_f32_16x16x32_bf16(
            false, qA[f], false, kB, (short)0, a, false, false);
      }
      const bool mz = (mask[b * SEQ + key] == 0);
#pragma unroll
      for (int j = 0; j < 8; ++j) a[j] = mz ? -1e9f : a[j] * 0.03125f; // 1/sqrt(E)
      sc[nt] = a;
    }

    // online softmax (rows live in the lane's 16-lane half-group)
    float bm[8];
#pragma unroll
    for (int j = 0; j < 8; ++j) bm[j] = fmaxf(sc[0][j], sc[1][j]);
#pragma unroll
    for (int off = 1; off < 16; off <<= 1)
#pragma unroll
      for (int j = 0; j < 8; ++j) bm[j] = fmaxf(bm[j], __shfl_xor(bm[j], off, 32));

    float corr[8], rs[8];
    v8f p0, p1;
#pragma unroll
    for (int j = 0; j < 8; ++j) {
      const float nm = fmaxf(mrun[j], bm[j]);
      corr[j] = __expf(mrun[j] - nm);
      mrun[j] = nm;
      p0[j] = __expf(sc[0][j] - nm);
      p1[j] = __expf(sc[1][j] - nm);
      rs[j] = p0[j] + p1[j];
    }
#pragma unroll
    for (int off = 1; off < 16; off <<= 1)
#pragma unroll
      for (int j = 0; j < 8; ++j) rs[j] += __shfl_xor(rs[j], off, 32);
#pragma unroll
    for (int j = 0; j < 8; ++j) lrun[j] = lrun[j] * corr[j] + rs[j];
#pragma unroll
    for (int i = 0; i < 4; ++i)
#pragma unroll
      for (int j = 0; j < 8; ++j) accv[i][j] *= corr[j];

    // transpose P (D-layout -> A-layout) through LDS
#pragma unroll
    for (int j = 0; j < 8; ++j) {
      pS[w][half * 8 + j][lr]      = f2bf(p0[j]);
      pS[w][half * 8 + j][16 + lr] = f2bf(p1[j]);
    }
    __syncthreads();

    // ctx += P(16x32) x V(32x64): 4 WMMAs
    const __bf16* pp = &pS[w][lr][half * 8];
    v16bf pA = make_frag(pp, pp + 16);
#pragma unroll
    for (int dt = 0; dt < 4; ++dt) {
      const __bf16* vp = &vT[dt * 16 + lr][half * 8];
      v16bf vB = make_frag(vp, vp + 16);
      accv[dt] = __builtin_amdgcn_wmma_f32_16x16x32_bf16(
          false, pA, false, vB, (short)0, accv[dt], false, false);
    }
  }

  // finalize: divide by row sums, store ctx bf16 (head-concat layout [B,S,E])
#pragma unroll
  for (int dt = 0; dt < 4; ++dt)
#pragma unroll
    for (int j = 0; j < 8; ++j) {
      const int gm = q0 + half * 8 + j;
      ctx[(size_t)(b * SEQ + gm) * EMB + h * HDIM + dt * 16 + lr] =
          f2bf(accv[dt][j] / lrun[j]);
    }
}

// ---------------- host orchestration ----------------
extern "C" void kernel_launch(void* const* d_in, const int* in_sizes, int n_in,
                              void* d_out, int out_size, void* d_ws, size_t ws_size,
                              hipStream_t stream) {
  (void)in_sizes; (void)n_in; (void)out_size; (void)ws_size;
  const float* I      = (const float*)d_in[0];
  const float* x      = (const float*)d_in[1];
  const int*   mask   = (const int*)d_in[2];
  const float* wq     = (const float*)d_in[3];
  const float* bq     = (const float*)d_in[4];
  const float* wk     = (const float*)d_in[5];
  const float* bk     = (const float*)d_in[6];
  const float* wv     = (const float*)d_in[7];
  const float* bv     = (const float*)d_in[8];
  const float* w_proj = (const float*)d_in[9];
  const float* b_proj = (const float*)d_in[10];
  const float* g1     = (const float*)d_in[11];
  const float* be1    = (const float*)d_in[12];
  const float* g2     = (const float*)d_in[13];
  const float* be2    = (const float*)d_in[14];
  const float* w1     = (const float*)d_in[15];
  const float* b1     = (const float*)d_in[16];
  const float* w2     = (const float*)d_in[17];
  const float* b2     = (const float*)d_in[18];

  const long nBSE = (long)MROWS * EMB;        // 4,194,304
  const long nEE  = (long)EMB * EMB;          // 1,048,576
  const long nEF  = (long)EMB * FF;           // 4,194,304

  char* wsp = (char*)d_ws;
  auto take = [&](size_t bytes) -> void* {
    void* p = (void*)wsp;
    wsp += (bytes + 255) & ~(size_t)255;
    return p;
  };
  float*  xnF   = (float*)take(nBSE * 4);
  __bf16* xnB   = (__bf16*)take(nBSE * 2);
  __bf16* Ib    = (__bf16*)take(nBSE * 2);
  __bf16* wqb   = (__bf16*)take(nEE * 2);
  __bf16* wkb   = (__bf16*)take(nEE * 2);
  __bf16* wvb   = (__bf16*)take(nEE * 2);
  __bf16* wpb   = (__bf16*)take(nEE * 2);
  __bf16* w1b   = (__bf16*)take(nEF * 2);
  __bf16* w2b   = (__bf16*)take(nEF * 2);
  __bf16* qb    = (__bf16*)take(nBSE * 2);
  __bf16* kbuf  = (__bf16*)take(nBSE * 2);
  __bf16* vbuf  = (__bf16*)take(nBSE * 2);
  __bf16* ctxb  = (__bf16*)take(nBSE * 2);
  float*  hF    = (float*)take(nBSE * 4);
  float*  h2F   = (float*)take(nBSE * 4);
  __bf16* h2B   = (__bf16*)take(nBSE * 2);
  __bf16* ffmid = (__bf16*)take((long)MROWS * FF * 2);

  // 1) precision conversion / weight relayout
  cvt_f32_bf16<<<2048, 256, 0, stream>>>(I, Ib, nBSE);
  cvt_qkv_w<<<1024, 256, 0, stream>>>(wq, wqb);
  cvt_qkv_w<<<1024, 256, 0, stream>>>(wk, wkb);
  cvt_qkv_w<<<1024, 256, 0, stream>>>(wv, wvb);
  cvt_f32_bf16<<<1024, 256, 0, stream>>>(w_proj, wpb, nEE);
  cvt_f32_bf16<<<2048, 256, 0, stream>>>(w1, w1b, nEF);
  cvt_f32_bf16<<<2048, 256, 0, stream>>>(w2, w2b, nEF);

  // 2) LN1
  ln_kernel<<<MROWS, 256, 0, stream>>>(x, g1, be1, xnF, xnB);

  // 3) Q/K/V projections (M=4096, N=1024, K=1024)
  {
    dim3 g(MROWS / BM, EMB / BN);
    gemm_bf16<<<g, 256, 0, stream>>>(Ib,  wqb, bq, nullptr, nullptr, qb,   MROWS, EMB, EMB, 0);
    gemm_bf16<<<g, 256, 0, stream>>>(xnB, wkb, bk, nullptr, nullptr, kbuf, MROWS, EMB, EMB, 0);
    gemm_bf16<<<g, 256, 0, stream>>>(xnB, wvb, bv, nullptr, nullptr, vbuf, MROWS, EMB, EMB, 0);
  }

  // 4) attention
  {
    dim3 g(SEQ / 64, HEADS, BATCH);
    attn_kernel<<<g, 128, 0, stream>>>(qb, kbuf, vbuf, mask, ctxb);
  }

  // 5) output projection + residual (h = xn + ctx@Wp + bp), fp32 out
  {
    dim3 g(MROWS / BM, EMB / BN);
    gemm_bf16<<<g, 256, 0, stream>>>(ctxb, wpb, b_proj, xnF, hF, nullptr, MROWS, EMB, EMB, 0);
  }

  // 6) LN2
  ln_kernel<<<MROWS, 256, 0, stream>>>(hF, g2, be2, h2F, h2B);

  // 7) FFN1 with exact GELU (M=4096, N=4096, K=1024)
  {
    dim3 g(MROWS / BM, FF / BN);
    gemm_bf16<<<g, 256, 0, stream>>>(h2B, w1b, b1, nullptr, nullptr, ffmid, MROWS, FF, EMB, 1);
  }

  // 8) FFN2 + residual -> final output (M=4096, N=1024, K=4096), fp32
  {
    dim3 g(MROWS / BM, EMB / BN);
    gemm_bf16<<<g, 256, 0, stream>>>(ffmid, w2b, b2, h2F, (float*)d_out, nullptr,
                                     MROWS, EMB, FF, 0);
  }
}
